// PharmEncoder_22368189678094
// MI455X (gfx1250) — matrix operand order repaired and checked by reading, not compiled, not run on hardware
//
#include <hip/hip_runtime.h>
#include <cstring>
#include <cmath>

// ---------------------------------------------------------------------------
// MVMP / PharmEncoder pipeline for MI455X (gfx1250, wave32, WMMA).
// GEMMs run in bf16 via v_wmma_f32_16x16x32_bf16 with f32 accumulation.
// D = 300 padded to 320 (10 K-chunks of 32, 20 N-tiles of 16).
// Weights are pre-swizzled into the WMMA B-fragment layout; LDS is
// double-buffered so the K-loop has a single barrier per chunk.
// ---------------------------------------------------------------------------

#define DIM  300
#define DP   320
#define NKCH 10     // 320 / 32 K-chunks
#define NTT  20     // 320 / 16 N-tiles total
#define DK   75     // head dim
#define NH   4      // heads

typedef __attribute__((ext_vector_type(16))) __bf16 v16bf;
typedef __attribute__((ext_vector_type(8)))  float  v8f;

__device__ __forceinline__ unsigned short f2bf(float f) {
  unsigned int u = __float_as_uint(f);
  u += 0x7FFFu + ((u >> 16) & 1u);           // round-to-nearest-even
  return (unsigned short)(u >> 16);
}
__device__ __forceinline__ int f_ord(float f) {   // monotone float->int map
  int i = __float_as_int(f);
  return (i >= 0) ? i : (i ^ 0x7FFFFFFF);
}
__device__ __forceinline__ float f_unord(int i) {
  return __int_as_float((i >= 0) ? i : (i ^ 0x7FFFFFFF));
}

// ---------------------------------------------------------------------------
// WMMA GEMM:  C[M x 300] (f32) = A[M x 320] (bf16) * Bp (swizzled bf16 panel)
//             (+ bias[col]) (+= C if accum)
// Block: 256 threads = 8 waves; tile 128(M) x 64(N); wave = 16 x 64 outputs.
// Bp layout (uints): [kc][ntile][g][n][8]  -- per-lane fragment stream.
// Double-buffered LDS: one barrier per K-chunk; global loads for chunk k+1
// issued before the barrier, stores overlap with WMMA issue.
// ---------------------------------------------------------------------------
#define AS_STRIDE 40   // ushorts; 80B rows: 16B-aligned b128, banks 20*ln distinct
#define BSLOT     12   // uints per lane slot (8 used + 4 pad): 48B, conflict-free

__global__ __launch_bounds__(256)
void gemm_bf16_320(const unsigned short* __restrict__ A,
                   const unsigned int* __restrict__ Bp,
                   float* __restrict__ C, int M,
                   const float* __restrict__ bias, int accum)
{
  __shared__ unsigned short As[2][128 * AS_STRIDE];   // 2 x 10240 B
  __shared__ unsigned int   Bs[2][128 * BSLOT];       // 2 x  6144 B

  const int tid  = threadIdx.x;
  const int lane = tid & 31;
  const int wave = tid >> 5;      // 8 waves -> 8 M sub-tiles of 16
  const int g    = lane >> 4;     // half-wave group
  const int ln   = lane & 15;
  const int m0   = blockIdx.x * 128;
  const int n0   = blockIdx.y * 64;
  const int nb0  = blockIdx.y * 4;   // first n-tile of this block

  v8f acc[4];
#pragma unroll
  for (int nt = 0; nt < 4; ++nt)
    acc[nt] = (v8f){0.f,0.f,0.f,0.f,0.f,0.f,0.f,0.f};

  const int arow  = tid >> 1;          // 0..127
  const int ahalf = (tid & 1) * 16;    // 16-ushort half (32B)
  const int bslot = tid >> 1;          // 0..127  (nt,g,n) slot
  const int bpart = (tid & 1) * 4;     // 4-uint half of the 8-uint payload
  const int gr    = m0 + arow;

  uint4 ra0, ra1, rb;

  // ---- prologue: stage chunk 0 into buffer 0 ----
  {
    ra0 = make_uint4(0u,0u,0u,0u); ra1 = ra0;
    if (gr < M) {
      const uint4* ap = (const uint4*)(A + (size_t)gr * DP + ahalf);
      ra0 = ap[0];
      ra1 = ap[1];
    }
    rb = *(const uint4*)(Bp + ((size_t)nb0 << 8) + tid * 4);
    uint4* asd = (uint4*)(&As[0][arow * AS_STRIDE + ahalf]);
    asd[0] = ra0;
    asd[1] = ra1;
    *(uint4*)(&Bs[0][bslot * BSLOT + bpart]) = rb;
  }

  for (int kc = 0; kc < NKCH; ++kc) {
    const int cur = kc & 1;

    // ---- issue global loads for chunk kc+1 (latency hidden by compute) ----
    if (kc + 1 < NKCH) {
      ra0 = make_uint4(0u,0u,0u,0u); ra1 = ra0;
      if (gr < M) {
        const uint4* ap =
            (const uint4*)(A + (size_t)gr * DP + (kc + 1) * 32 + ahalf);
        ra0 = ap[0];
        ra1 = ap[1];
      }
      rb = *(const uint4*)(Bp + ((size_t)((kc + 1) * NTT + nb0) << 8) + tid * 4);
    }

    __syncthreads();   // buffer `cur` fully staged; prior reads of cur^1 done

    // ---- A fragment (16x32): 2 contiguous b128 LDS loads ----
    union { v16bf v; unsigned int u[8]; } af;
    const unsigned short* apr = &As[cur][(wave * 16 + ln) * AS_STRIDE];
#pragma unroll
    for (int i = 0; i < 8; ++i) {
      const int k = (g << 4) + (i << 1);
      af.u[i] = *(const unsigned int*)(apr + k);
    }

    // ---- 4 B fragments (pre-swizzled) + 4 WMMAs, A reused ----
#pragma unroll
    for (int nt = 0; nt < 4; ++nt) {
      union { v16bf v; unsigned int u[8]; } bfg;
      const unsigned int* bpr = &Bs[cur][(((nt << 1) + g) * 16 + ln) * BSLOT];
#pragma unroll
      for (int i = 0; i < 8; ++i) bfg.u[i] = bpr[i];
      acc[nt] = __builtin_amdgcn_wmma_f32_16x16x32_bf16(
          false, af.v, false, bfg.v, (short)0, acc[nt], false, false);
    }

    // ---- store chunk kc+1 into the other buffer (no barrier needed) ----
    if (kc + 1 < NKCH) {
      uint4* asd = (uint4*)(&As[cur ^ 1][arow * AS_STRIDE + ahalf]);
      asd[0] = ra0;
      asd[1] = ra1;
      *(uint4*)(&Bs[cur ^ 1][bslot * BSLOT + bpart]) = rb;
    }
  }

  // ---- epilogue: C/D layout -> M = r + 8*g, N = ln ----
#pragma unroll
  for (int nt = 0; nt < 4; ++nt) {
    const int col = n0 + nt * 16 + ln;
    if (col >= DIM) continue;
#pragma unroll
    for (int r = 0; r < 8; ++r) {
      const int row = m0 + wave * 16 + r + (g << 3);
      if (row >= M) continue;
      float v = acc[nt][r];
      if (bias)  v += bias[col];
      if (accum) v += C[(size_t)row * DIM + col];
      C[(size_t)row * DIM + col] = v;
    }
  }
}

// ---------------------------------------------------------------------------
// Elementwise / scatter kernels
// ---------------------------------------------------------------------------
__global__ void k_fill_f32(float* p, float v, int n) {
  int i = blockIdx.x * 256 + threadIdx.x;
  if (i < n) p[i] = v;
}
__global__ void k_fill_i32(int* p, int v, int n) {
  int i = blockIdx.x * 256 + threadIdx.x;
  if (i < n) p[i] = v;
}
__global__ void k_deg(const int* __restrict__ dst, int E, float* deg) {
  int e = blockIdx.x * 256 + threadIdx.x;
  if (e < E) atomicAdd(&deg[dst[e]], 1.0f);
}
// f32 [M x 300] -> bf16 [M x 320] (zero pad), optional f32 copy-out
__global__ void k_conv_act(const float* __restrict__ src, float* fcopy,
                           unsigned short* __restrict__ dstb, int M) {
  int i = blockIdx.x * 256 + threadIdx.x;
  if (i >= M * DP) return;
  int r = i / DP, c = i % DP;
  float v = (c < DIM) ? src[(size_t)r * DIM + c] : 0.f;
  dstb[i] = f2bf(v);
  if (fcopy && c < DIM) fcopy[(size_t)r * DIM + c] = v;
}
// weight f32 [300 x 300] -> swizzled bf16 panel [kc][nt][g][n][8 uints]
__global__ void k_conv_w_swz(const float* __restrict__ src,
                             unsigned int* __restrict__ dst) {
  int idx = blockIdx.x * 256 + threadIdx.x;       // 10*20*2*16*8 = 51200
  if (idx >= NKCH * NTT * 256) return;
  const int i8   = idx & 7;
  const int slot = idx >> 3;
  const int n    = slot & 15;
  const int gsl  = slot >> 4;
  const int g    = gsl & 1;
  const int t2   = gsl >> 1;
  const int nt   = t2 % NTT;
  const int kc   = t2 / NTT;
  const int k    = kc * 32 + g * 16 + 2 * i8;
  const int col  = nt * 16 + n;
  float v0 = 0.f, v1 = 0.f;
  if (col < DIM) {
    if (k     < DIM) v0 = src[k * DIM + col];
    if (k + 1 < DIM) v1 = src[(k + 1) * DIM + col];
  }
  dst[idx] = (unsigned int)f2bf(v0) | ((unsigned int)f2bf(v1) << 16);
}
// per-(edge,head) score + segment max (ordered-int atomicMax)
__global__ void k_edge_score(const float* __restrict__ q,
                             const float* __restrict__ kk,
                             const int* __restrict__ dst, int E, float scale,
                             float* __restrict__ sbuf, int* __restrict__ smax) {
  int i = blockIdx.x * 256 + threadIdx.x;
  if (i >= E * NH) return;
  int e = i >> 2, h = i & 3;
  const float* qr = q  + (size_t)dst[e] * DIM + h * DK;
  const float* kr = kk + (size_t)e      * DIM + h * DK;
  float s = 0.f;
#pragma unroll 5
  for (int j = 0; j < DK; ++j) s = fmaf(qr[j], kr[j], s);
  s *= scale;
  sbuf[i] = s;
  atomicMax(&smax[dst[e] * NH + h], f_ord(s));
}
__global__ void k_edge_p(const float* __restrict__ sbuf,
                         const int* __restrict__ smax,
                         const int* __restrict__ dst, int E,
                         float* __restrict__ pbuf, float* __restrict__ Zb) {
  int i = blockIdx.x * 256 + threadIdx.x;
  if (i >= E * NH) return;
  int e = i >> 2, h = i & 3;
  float sm = f_unord(smax[dst[e] * NH + h]);
  float p = __expf(sbuf[i] - sm);
  pbuf[i] = p;
  atomicAdd(&Zb[dst[e] * NH + h], p);
}
__global__ void k_ctx_scatter(const float* __restrict__ pbuf,
                              const float* __restrict__ v,
                              const int* __restrict__ dst, int E,
                              float* __restrict__ ctx) {
  int i = blockIdx.x * 256 + threadIdx.x;
  if (i >= E * DIM) return;
  int e = i / DIM, d = i % DIM;
  float pv = pbuf[e * NH + d / DK];
  atomicAdd(&ctx[(size_t)dst[e] * DIM + d], pv * v[i]);
}
__global__ void k_ctx_norm(float* __restrict__ ctx, const float* __restrict__ Zb,
                           int N) {
  int i = blockIdx.x * 256 + threadIdx.x;
  if (i >= N * DIM) return;
  int n = i / DIM, d = i % DIM;
  float Zv = Zb[n * NH + d / DK];
  if (Zv > 0.f) ctx[i] /= Zv;
}
// fh = has_msg ? attn_out + fh : fh ; refresh bf16 mirror
__global__ void k_fh_update(const float* __restrict__ o,
                            const float* __restrict__ deg, int N,
                            float* __restrict__ fh,
                            unsigned short* __restrict__ fhb) {
  int i = blockIdx.x * 256 + threadIdx.x;
  if (i >= N * DIM) return;
  int n = i / DIM, d = i % DIM;
  float v = fh[i];
  if (deg[n] > 0.f) v += o[i];
  fh[i] = v;
  fhb[(size_t)n * DP + d] = f2bf(v);
}
// m = fh[src] - h[e^1]  (reverse-pair edges), emitted directly as padded bf16
__global__ void k_edge_m(const float* __restrict__ fh,
                         const int* __restrict__ src,
                         const float* __restrict__ h, int E,
                         unsigned short* __restrict__ mbuf) {
  int i = blockIdx.x * 256 + threadIdx.x;
  if (i >= E * DP) return;
  int e = i / DP, c = i % DP;
  float v = 0.f;
  if (c < DIM)
    v = fh[(size_t)src[e] * DIM + c] - h[(size_t)(e ^ 1) * DIM + c];
  mbuf[i] = f2bf(v);
}
// h = relu(x + (m @ W + b)) ; refresh bf16 mirror
__global__ void k_h_update(const float* __restrict__ x, int E,
                           float* __restrict__ h,
                           unsigned short* __restrict__ hb) {
  int i = blockIdx.x * 256 + threadIdx.x;
  if (i >= E * DIM) return;
  int e = i / DIM, d = i % DIM;
  float v = x[i] + h[i];
  v = v > 0.f ? v : 0.f;
  h[i] = v;
  hb[(size_t)e * DP + d] = f2bf(v);
}
__global__ void k_mail_scatter(const float* __restrict__ h,
                               const int* __restrict__ dst, int E,
                               float* __restrict__ mail) {
  int i = blockIdx.x * 256 + threadIdx.x;
  if (i >= E * DIM) return;
  int e = i / DIM, d = i % DIM;
  atomicAdd(&mail[(size_t)dst[e] * DIM + d], h[i]);
}
__global__ void k_blend(float* __restrict__ out, const float* __restrict__ fh,
                        const float* __restrict__ deg, int N) {
  int i = blockIdx.x * 256 + threadIdx.x;
  if (i >= N * DIM) return;
  if (deg[i / DIM] <= 0.f) out[i] = fh[i];
}

// ---------------------------------------------------------------------------
// Host-side branch driver
// ---------------------------------------------------------------------------
#define L1(kern, n, ...) \
  kern<<<dim3(((n) + 255) / 256), dim3(256), 0, stream>>>(__VA_ARGS__)

static inline void gemm(const unsigned short* A, const unsigned int* Bp,
                        float* C, int M, const float* bias, int acc,
                        hipStream_t stream) {
  dim3 grid((M + 127) / 128, 5);
  gemm_bf16_320<<<grid, 256, 0, stream>>>(A, Bp, C, M, bias, acc);
}

static void run_branch(const float* f, const float* x, const float* attn_w,
                       const float* attn_b, const float* mp_w,
                       const float* mp_b, const float* last_w,
                       const float* last_b, const int* src, const int* dst,
                       int N, int E, float* out,
                       unsigned short* hb, unsigned short* mb,
                       unsigned short* fb, unsigned short* fhb,
                       unsigned int* const* W,
                       float* qbuf, float* kbuf, float* vbuf, float* hf,
                       float* fhf, float* sbuf, float* pbuf, float* Zb,
                       float* degb, int* smax, hipStream_t stream) {
  const float scale = 1.0f / sqrtf((float)DK);
  int ordm;
  {
    float ninf = -INFINITY; int b; memcpy(&b, &ninf, 4);
    ordm = (b >= 0) ? b : (b ^ 0x7FFFFFFF);
  }
  const int WGRID = (NKCH * NTT * 256) / 256;   // conv_w_swz blocks

  // setup: degree mask, bf16 mirrors, weight panels
  L1(k_fill_f32, N, degb, 0.f, N);
  L1(k_deg, E, dst, E, degb);
  L1(k_conv_act, N * DP, f, fhf, fhb, N);             // fh = f (+bf16)
  L1(k_conv_act, N * DP, f, (float*)nullptr, fb, N);  // f bf16 (const)
  L1(k_conv_act, E * DP, x, hf, hb, E);               // h = x (+bf16)
  for (int w = 0; w < 4; ++w)
    k_conv_w_swz<<<dim3(WGRID), 256, 0, stream>>>(attn_w + w * DIM * DIM, W[w]);
  for (int w = 0; w < 2; ++w)
    k_conv_w_swz<<<dim3(WGRID), 256, 0, stream>>>(mp_w + w * DIM * DIM, W[4 + w]);
  for (int w = 0; w < 3; ++w)
    k_conv_w_swz<<<dim3(WGRID), 256, 0, stream>>>(last_w + w * DIM * DIM, W[6 + w]);

  for (int it = 0; it < 2; ++it) {
    // attention projections (WMMA GEMMs)
    gemm(fhb, W[0], qbuf, N, attn_b + 0 * DIM, 0, stream);  // q
    gemm(hb,  W[1], kbuf, E, attn_b + 1 * DIM, 0, stream);  // k
    gemm(hb,  W[2], vbuf, E, attn_b + 2 * DIM, 0, stream);  // v
    // segment softmax over incoming edges
    L1(k_fill_i32, N * NH, smax, ordm, N * NH);
    L1(k_fill_f32, N * NH, Zb, 0.f, N * NH);
    L1(k_edge_score, E * NH, qbuf, kbuf, dst, E, scale, sbuf, smax);
    L1(k_edge_p, E * NH, sbuf, smax, dst, E, pbuf, Zb);
    L1(k_fill_f32, N * DIM, qbuf, 0.f, N * DIM);            // ctx reuses qbuf
    L1(k_ctx_scatter, E * DIM, pbuf, vbuf, dst, E, qbuf);
    L1(k_ctx_norm, N * DIM, qbuf, Zb, N);
    // output projection + residual
    L1(k_conv_act, N * DP, qbuf, (float*)nullptr, mb, N);   // ctx -> bf16
    gemm(mb, W[3], kbuf, N, attn_b + 3 * DIM, 0, stream);   // o reuses kbuf
    L1(k_fh_update, N * DIM, kbuf, degb, N, fhf, fhb);
    // edge update: h = relu(x + (fh[src] - rev_h) @ mp_w + b)
    L1(k_edge_m, E * DP, fhf, src, hf, E, mb);
    gemm(mb, W[4 + it], hf, E, mp_b + it * DIM, 0, stream);
    L1(k_h_update, E * DIM, x, E, hf, hb);
  }

  // mail = segment_sum(h); new = [mail, fh, f] @ last_w + last_b
  L1(k_fill_f32, N * DIM, qbuf, 0.f, N * DIM);              // mail reuses qbuf
  L1(k_mail_scatter, E * DIM, hf, dst, E, qbuf);
  L1(k_conv_act, N * DP, qbuf, (float*)nullptr, mb, N);
  gemm(mb,  W[6], out, N, last_b, 0, stream);
  gemm(fhb, W[7], out, N, nullptr, 1, stream);
  gemm(fb,  W[8], out, N, nullptr, 1, stream);
  L1(k_blend, N * DIM, out, fhf, degb, N);
}

// ---------------------------------------------------------------------------
extern "C" void kernel_launch(void* const* d_in, const int* in_sizes, int n_in,
                              void* d_out, int out_size, void* d_ws,
                              size_t ws_size, hipStream_t stream) {
  const float* f_a    = (const float*)d_in[0];
  const float* x_a    = (const float*)d_in[1];
  const float* f_p    = (const float*)d_in[2];
  const float* x_p    = (const float*)d_in[3];
  const float* aw_a   = (const float*)d_in[4];
  const float* ab_a   = (const float*)d_in[5];
  const float* aw_p   = (const float*)d_in[6];
  const float* ab_p   = (const float*)d_in[7];
  const float* mw_a   = (const float*)d_in[8];
  const float* mb_a   = (const float*)d_in[9];
  const float* mw_p   = (const float*)d_in[10];
  const float* mb_p   = (const float*)d_in[11];
  const float* lw_a   = (const float*)d_in[12];
  const float* lb_a   = (const float*)d_in[13];
  const float* lw_p   = (const float*)d_in[14];
  const float* lb_p   = (const float*)d_in[15];
  const int*   src_a  = (const int*)d_in[16];
  const int*   dst_a  = (const int*)d_in[17];
  const int*   src_p  = (const int*)d_in[18];
  const int*   dst_p  = (const int*)d_in[19];

  const int Na = in_sizes[0] / DIM, Ea = in_sizes[1] / DIM;
  const int Np = in_sizes[2] / DIM, Ep = in_sizes[3] / DIM;
  const int Nmax = Na > Np ? Na : Np;
  const int Emax = Ea > Ep ? Ea : Ep;

  // bump allocator over workspace
  char* basep = (char*)d_ws;
  size_t off = 0;
  auto alloc = [&](size_t bytes) -> void* {
    void* p = basep + off;
    off = (off + bytes + 255) & ~(size_t)255;
    return p;
  };
  unsigned short* hb  = (unsigned short*)alloc((size_t)Emax * DP * 2);
  unsigned short* mb  = (unsigned short*)alloc((size_t)Emax * DP * 2);
  unsigned short* fb  = (unsigned short*)alloc((size_t)Nmax * DP * 2);
  unsigned short* fhb = (unsigned short*)alloc((size_t)Nmax * DP * 2);
  unsigned int* W[9];
  for (int i = 0; i < 9; ++i) W[i] = (unsigned int*)alloc((size_t)DP * DP * 2);
  float* qbuf = (float*)alloc((size_t)Nmax * DIM * 4);
  float* kbuf = (float*)alloc((size_t)Emax * DIM * 4);
  float* vbuf = (float*)alloc((size_t)Emax * DIM * 4);
  float* hf   = (float*)alloc((size_t)Emax * DIM * 4);
  float* fhf  = (float*)alloc((size_t)Nmax * DIM * 4);
  float* sbuf = (float*)alloc((size_t)Emax * NH * 4);
  float* pbuf = (float*)alloc((size_t)Emax * NH * 4);
  float* Zb   = (float*)alloc((size_t)Nmax * NH * 4);
  float* degb = (float*)alloc((size_t)Nmax * 4);
  int*   smax = (int*)alloc((size_t)Nmax * NH * 4);

  float* out_a = (float*)d_out;
  float* out_p = out_a + (size_t)Na * DIM;

  run_branch(f_a, x_a, aw_a, ab_a, mw_a, mb_a, lw_a, lb_a, src_a, dst_a,
             Na, Ea, out_a, hb, mb, fb, fhb, W, qbuf, kbuf, vbuf, hf, fhf,
             sbuf, pbuf, Zb, degb, smax, stream);
  run_branch(f_p, x_p, aw_p, ab_p, mw_p, mb_p, lw_p, lb_p, src_p, dst_p,
             Np, Ep, out_p, hb, mb, fb, fhb, W, qbuf, kbuf, vbuf, hf, fhf,
             sbuf, pbuf, Zb, degb, smax, stream);
}
